// Net_43361989821037
// MI455X (gfx1250) — matrix operand order repaired and checked by reading, not compiled, not run on hardware
//
#include <hip/hip_runtime.h>
#include <hip/hip_bf16.h>
#include <math.h>

// ---------------- model constants ----------------
#define DM   300      // d_model
#define NH   2        // heads
#define DH   150      // d_model / heads
#define DFF  1024
#define HID  128
#define OUTC 2
#define BB   16       // batch
#define LL   512      // seq len (layer 0)
#define LT   514      // conv output length (wrap-pad 2, VALID k=3)
#define LP   257      // pooled length (pad 1 -inf, win 3 stride 2)
#define U1   35       // 5*ceil(ln 512)
#define U2   30       // 5*ceil(ln 257)

#define NP300   320   // DM (=300) padded to 64-multiple for N blocking
#define KP_DM   320   // DM padded to 32 (K dim)
#define KP_CONV 928   // 900 padded to 32
#define KP_FF   1024

typedef __attribute__((ext_vector_type(16))) _Float16 v16h;
typedef __attribute__((ext_vector_type(8)))  _Float16 v8h;
typedef __attribute__((ext_vector_type(8)))  float    v8f;

#define NEG_INF (-3.402823466e38f)

// =====================================================================
// f32 -> padded f16 staging (zero-fills K and row padding).
// =====================================================================
__global__ void cvt_pad_f16(const float* __restrict__ src, _Float16* __restrict__ dst,
                            int rows, int K, int Kp, int rowsPad)
{
    const long total = (long)rowsPad * Kp;
    for (long i = (long)blockIdx.x * blockDim.x + threadIdx.x; i < total;
         i += (long)gridDim.x * blockDim.x) {
        const long r = i / Kp;
        const int  k = (int)(i - r * Kp);
        float v = (r < rows && k < K) ? src[r * (long)K + k] : 0.0f;
        dst[i] = (_Float16)v;
    }
}

// ---- fragment loaders (two 16-B vector loads each, layout per ISA §7.12.2) ----
__device__ __forceinline__ v16h load_fragA(const _Float16* arow, int k0)
{
    v8h a0 = *(const v8h*)(arow + k0);
    v8h a1 = *(const v8h*)(arow + k0 + 16);
    return __builtin_shufflevector(a0, a1, 0,1,2,3,4,5,6,7,8,9,10,11,12,13,14,15);
}
__device__ __forceinline__ v16h load_fragB(const _Float16* brow, int k0)
{
    v8h b0 = *(const v8h*)(brow + k0);
    v8h b1 = *(const v8h*)(brow + k0 + 16);
    return __builtin_shufflevector(b0, b1, 0,1,2,3,4,5,6,7,8,9,10,11,12,13,14,15);
}

// =====================================================================
// WMMA GEMM on padded f16 operands, 16x64 strip per wave (4 N-tiles):
//   C[M,N] = act(A[M,Kp] @ W[Npad,Kp]^T + bias + R)
// M multiple of 16, Kp multiple of 32, Npad multiple of 64 (W zero-padded).
// Software-pipelined: chunk k+1's fragments are loaded before chunk k's
// 4 WMMAs issue, so the matrix ops only wait on one-iteration-old loads.
// ACT: 0 = none, 1 = exact GELU
// =====================================================================
template<int ACT>
__global__ void wmma_gemm_f16(const _Float16* __restrict__ A,
                              const _Float16* __restrict__ W,
                              const float* __restrict__ bias,
                              const float* __restrict__ R,
                              float* __restrict__ C,
                              int M, int N, int Npad, int Kp)
{
    const int tilesN4 = Npad >> 6;                // groups of 4 N-tiles
    const int tilesM  = M >> 4;
    const int wave    = (blockIdx.x * blockDim.x + threadIdx.x) >> 5;
    if (wave >= tilesM * tilesN4) return;         // wave-uniform exit

    const int lane    = threadIdx.x & 31;
    const int halfSel = lane >> 4;                // 0 or 1
    const int l16     = lane & 15;

    const int tm   = wave / tilesN4;
    const int tn   = wave % tilesN4;
    const int row0 = tm << 4;
    const int col0 = tn << 6;

    const _Float16* arow = A + (size_t)(row0 + l16) * Kp + (halfSel << 3);
    const _Float16* brow = W + (size_t)(col0 + l16) * Kp + (halfSel << 3);
    const size_t bstep = (size_t)16 * Kp;

    v8f acc[4];
#pragma unroll
    for (int j = 0; j < 4; ++j) acc[j] = (v8f){0.f,0.f,0.f,0.f,0.f,0.f,0.f,0.f};

    // prologue: chunk 0 fragments
    v16h a = load_fragA(arow, 0);
    v16h b[4];
#pragma unroll
    for (int j = 0; j < 4; ++j) b[j] = load_fragB(brow + (size_t)j * bstep, 0);

    for (int k0 = 0; k0 < Kp; k0 += 32) {
        const int kn = k0 + 32;
        v16h an;
        v16h bn[4];
        if (kn < Kp) {                            // issue next-chunk loads first
            __builtin_prefetch(arow + kn + 96, 0, 1);   // global_prefetch_b8
            an = load_fragA(arow, kn);
#pragma unroll
            for (int j = 0; j < 4; ++j) bn[j] = load_fragB(brow + (size_t)j * bstep, kn);
        }
#pragma unroll
        for (int j = 0; j < 4; ++j) {
            acc[j] = __builtin_amdgcn_wmma_f32_16x16x32_f16(
                /*neg_a=*/false, a, /*neg_b=*/false, b[j],
                /*c_mod=*/(short)0, acc[j], /*reuse_a=*/false, /*reuse_b=*/false);
        }
        a = an;
#pragma unroll
        for (int j = 0; j < 4; ++j) b[j] = bn[j];
    }

#pragma unroll
    for (int j = 0; j < 4; ++j) {
        const int n = col0 + (j << 4) + l16;
        if (n < N) {
            const float bv = bias ? bias[n] : 0.0f;
#pragma unroll
            for (int r = 0; r < 8; ++r) {
                const int m = row0 + r + (halfSel << 3);   // C/D layout: VGPR r -> M
                float v = acc[j][r] + bv;
                if (R) v += R[(size_t)m * N + n];
                if (ACT == 1)   // exact GELU
                    v = 0.5f * v * (1.0f + erff(v * 0.70710678118654752f));
                C[(size_t)m * N + n] = v;
            }
        }
    }
}

// Host-side: stage A and W to f16 then run the WMMA GEMM.
static void launch_cvt(const float* src, _Float16* dst, int rows, int K, int Kp, int rowsPad,
                       hipStream_t s)
{
    const long total = (long)rowsPad * Kp;
    const int blocks = (int)((total + 255) / 256);
    hipLaunchKernelGGL(cvt_pad_f16, dim3(blocks), dim3(256), 0, s, src, dst, rows, K, Kp, rowsPad);
}

static void launch_gemm(int act, const _Float16* Ah, const _Float16* Wh, const float* bias,
                        const float* R, float* C, int M, int N, int Npad, int Kp, hipStream_t s)
{
    const int tiles   = (M / 16) * (Npad / 64);    // one wave per 16x64 strip
    const int threads = 256;                       // 8 waves / block (wave32)
    const int blocks  = (tiles + 7) / 8;
    if (act == 1) hipLaunchKernelGGL((wmma_gemm_f16<1>), dim3(blocks), dim3(threads), 0, s,
                                     Ah, Wh, bias, R, C, M, N, Npad, Kp);
    else          hipLaunchKernelGGL((wmma_gemm_f16<0>), dim3(blocks), dim3(threads), 0, s,
                                     Ah, Wh, bias, R, C, M, N, Npad, Kp);
}

// =====================================================================
// Embedding gather
// =====================================================================
__global__ void embed_kernel(const int* __restrict__ x, const float* __restrict__ emb,
                             float* __restrict__ E, long ntok, int D)
{
    long total = ntok * D;
    for (long i = (long)blockIdx.x * blockDim.x + threadIdx.x; i < total;
         i += (long)gridDim.x * blockDim.x) {
        long t = i / D; int d = (int)(i - t * D);
        E[i] = emb[(size_t)x[t] * D + d];
    }
}

// =====================================================================
// Row LayerNorm (block per row, N=300)
// =====================================================================
__global__ void layernorm_rows(const float* __restrict__ X, const float* __restrict__ g,
                               const float* __restrict__ b, float* __restrict__ Y, int N)
{
    const long row = blockIdx.x;
    const float* x = X + row * (long)N;
    float* y = Y + row * (long)N;
    __shared__ float s1[256], s2[256];
    const int tid = threadIdx.x, bs = blockDim.x;
    float a = 0.f, q = 0.f;
    for (int i = tid; i < N; i += bs) { float v = x[i]; a += v; q += v * v; }
    s1[tid] = a; s2[tid] = q; __syncthreads();
    for (int st = 128; st > 0; st >>= 1) {
        if (tid < st) { s1[tid] += s1[tid + st]; s2[tid] += s2[tid + st]; }
        __syncthreads();
    }
    const float mean = s1[0] / N;
    const float var  = s2[0] / N - mean * mean;
    const float inv  = rsqrtf(var + 1e-5f);
    for (int i = tid; i < N; i += bs) y[i] = (x[i] - mean) * inv * g[i] + b[i];
}

// =====================================================================
// Prob-sparse attention (deterministic hash stands in for jax PRNG)
// =====================================================================
__global__ void attn_measure(const float* __restrict__ q, const float* __restrict__ k,
                             float* __restrict__ Mout, int L, int D, int u)
{
    const int bh = blockIdx.x;
    const float* qb = q + (size_t)bh * L * D;
    const float* kb = k + (size_t)bh * L * D;
    for (int l = threadIdx.x; l < L; l += blockDim.x) {
        float mx = NEG_INF, sm = 0.f;
        for (int j = 0; j < u; ++j) {
            unsigned h = (unsigned)l * 2654435761u + (unsigned)j * 2246822519u + 12345u;
            const int kj = (int)(h % (unsigned)L);
            float dot = 0.f;
            const float* qr = qb + (size_t)l * D;
            const float* kr = kb + (size_t)kj * D;
            for (int d = 0; d < D; ++d) dot += qr[d] * kr[d];
            mx = fmaxf(mx, dot); sm += dot;
        }
        Mout[(size_t)bh * L + l] = mx - sm / (float)L;
    }
}

__global__ void attn_topk(const float* __restrict__ Mm, int* __restrict__ mtop, int L, int u)
{
    __shared__ float tmp[LL];
    const int bh = blockIdx.x;
    for (int i = threadIdx.x; i < L; i += blockDim.x) tmp[i] = Mm[(size_t)bh * L + i];
    __syncthreads();
    if (threadIdx.x == 0) {
        for (int s = 0; s < u; ++s) {
            int best = 0; float bv = NEG_INF;
            for (int i = 0; i < L; ++i) if (tmp[i] > bv) { bv = tmp[i]; best = i; }
            mtop[bh * u + s] = best;
            tmp[best] = NEG_INF;
        }
    }
}

__global__ void attn_ctx_base(const float* __restrict__ v, float* __restrict__ ctx, int L, int D)
{
    const int bh = blockIdx.x;
    __shared__ float vs[DH + 2];
    const float* vb = v + (size_t)bh * L * D;
    for (int d = threadIdx.x; d < D; d += blockDim.x) {
        float s = 0.f;
        for (int l = 0; l < L; ++l) s += vb[(size_t)l * D + d];
        vs[d] = s;
    }
    __syncthreads();
    float* cb = ctx + (size_t)bh * L * D;
    const long total = (long)L * D;
    for (long i = threadIdx.x; i < total; i += blockDim.x) cb[i] = vs[i % D];
}

__global__ void attn_upd(const float* __restrict__ q, const float* __restrict__ k,
                         const float* __restrict__ v, const int* __restrict__ mtop,
                         float* __restrict__ ctx, int L, int D, int u, float scale)
{
    const int bh = blockIdx.x / u;
    const int i  = blockIdx.x % u;
    const int lq = mtop[bh * u + i];
    const float* qr = q + ((size_t)bh * L + lq) * D;
    const float* kb = k + (size_t)bh * L * D;
    const float* vb = v + (size_t)bh * L * D;
    __shared__ float sc[LL];
    __shared__ float red[256];
    const int tid = threadIdx.x, bs = blockDim.x;
    float lmax = NEG_INF;
    for (int l = tid; l < L; l += bs) {
        float dot = 0.f;
        const float* kr = kb + (size_t)l * D;
        for (int d = 0; d < D; ++d) dot += qr[d] * kr[d];
        dot *= scale; sc[l] = dot; lmax = fmaxf(lmax, dot);
    }
    red[tid] = lmax; __syncthreads();
    for (int st = 128; st > 0; st >>= 1) {
        if (tid < st) red[tid] = fmaxf(red[tid], red[tid + st]);
        __syncthreads();
    }
    const float mx = red[0]; __syncthreads();
    float ls = 0.f;
    for (int l = tid; l < L; l += bs) { float e = expf(sc[l] - mx); sc[l] = e; ls += e; }
    red[tid] = ls; __syncthreads();
    for (int st = 128; st > 0; st >>= 1) {
        if (tid < st) red[tid] += red[tid + st];
        __syncthreads();
    }
    const float inv = 1.0f / red[0];
    __syncthreads();
    for (int d = tid; d < D; d += bs) {
        float acc = 0.f;
        for (int l = 0; l < L; ++l) acc += sc[l] * vb[(size_t)l * D + d];
        ctx[((size_t)bh * L + lq) * D + d] = acc * inv;
    }
}

// =====================================================================
// Conv layer pieces
// =====================================================================
__global__ void im2col_wrap(const float* __restrict__ E, float* __restrict__ col,
                            int Bv, int Lin, int Cc)
{
    const int Lo = Lin + 2;
    const long total = (long)Bv * Lo * Cc * 3;
    for (long x = (long)blockIdx.x * blockDim.x + threadIdx.x; x < total;
         x += (long)gridDim.x * blockDim.x) {
        const int  dk = (int)(x % 3);
        long r = x / 3;
        const int  ci = (int)(r % Cc); r /= Cc;
        const int  t  = (int)(r % Lo);
        const int  b  = (int)(r / Lo);
        int src = t + dk - 2;
        src = ((src % Lin) + Lin) % Lin;                 // wrap pad
        col[x] = E[((size_t)b * Lin + src) * Cc + ci];
    }
}

__global__ void chan_norm_elu(float* __restrict__ Y, const float* __restrict__ g,
                              const float* __restrict__ be, int Mrows, int Cc)
{
    __shared__ float s1[256], s2[256];
    const int c = blockIdx.x, tid = threadIdx.x, bs = blockDim.x;
    float a = 0.f, q = 0.f;
    for (int m = tid; m < Mrows; m += bs) { float v = Y[(size_t)m * Cc + c]; a += v; q += v * v; }
    s1[tid] = a; s2[tid] = q; __syncthreads();
    for (int st = 128; st > 0; st >>= 1) {
        if (tid < st) { s1[tid] += s1[tid + st]; s2[tid] += s2[tid + st]; }
        __syncthreads();
    }
    const float mean = s1[0] / Mrows;
    const float var  = s2[0] / Mrows - mean * mean;
    const float inv  = rsqrtf(var + 1e-5f);
    const float gg = g[c], bb = be[c];
    for (int m = tid; m < Mrows; m += bs) {
        float v = (Y[(size_t)m * Cc + c] - mean) * inv * gg + bb;
        Y[(size_t)m * Cc + c] = (v > 0.f) ? v : expm1f(v);     // ELU(alpha=1)
    }
}

__global__ void maxpool3s2(const float* __restrict__ Y, float* __restrict__ Eo,
                           int Bv, int Lc, int Cc)
{
    const int Lo = (Lc + 2 - 3) / 2 + 1;
    const long total = (long)Bv * Lo * Cc;
    for (long x = (long)blockIdx.x * blockDim.x + threadIdx.x; x < total;
         x += (long)gridDim.x * blockDim.x) {
        const int c = (int)(x % Cc);
        long r = x / Cc;
        const int t = (int)(r % Lo);
        const int b = (int)(r / Lo);
        float m = NEG_INF;
        for (int dj = 0; dj < 3; ++dj) {
            const int j = 2 * t + dj - 1;                 // -inf pad of 1 each side
            if (j >= 0 && j < Lc) m = fmaxf(m, Y[((size_t)b * Lc + j) * Cc + c]);
        }
        Eo[x] = m;
    }
}

// =====================================================================
// RNN scan (tanh cell), single block, h double-buffered in LDS
// =====================================================================
__global__ void rnn_scan(const float* __restrict__ xp, const float* __restrict__ h0,
                         const float* __restrict__ whh, const float* __restrict__ bhh,
                         float* __restrict__ hT, int T)
{
    __shared__ float h[BB * HID];
    __shared__ float hn[BB * HID];
    const int tid = threadIdx.x, bs = blockDim.x;
    for (int i = tid; i < BB * HID; i += bs) h[i] = h0[i];
    __syncthreads();
    for (int t = 0; t < T; ++t) {
        for (int i = tid; i < BB * HID; i += bs) {
            const int b = i >> 7, j = i & (HID - 1);
            float acc = xp[((size_t)b * T + t) * HID + j] + bhh[j];
            const float* wr = whh + (size_t)j * HID;
            const float* hb = h + b * HID;
            for (int kk = 0; kk < HID; ++kk) acc += hb[kk] * wr[kk];
            hn[i] = tanhf(acc);
        }
        __syncthreads();
        for (int i = tid; i < BB * HID; i += bs) h[i] = hn[i];
        __syncthreads();
    }
    for (int i = tid; i < BB * HID; i += bs) hT[i] = h[i];
}

__global__ void logits_ls(const float* __restrict__ hT, const float* __restrict__ wl,
                          const float* __restrict__ bl, float* __restrict__ out)
{
    __shared__ float lg[BB][OUTC];
    const int tid = threadIdx.x;
    if (tid < BB * OUTC) {
        const int b = tid / OUTC, o = tid % OUTC;
        float acc = bl[o];
        for (int kk = 0; kk < HID; ++kk) acc += hT[b * HID + kk] * wl[o * HID + kk];
        lg[b][o] = acc;
    }
    __syncthreads();
    if (tid < BB * OUTC) {
        const int b = tid / OUTC, o = tid % OUTC;
        float m = fmaxf(lg[b][0], lg[b][1]);
        float lse = m + logf(expf(lg[b][0] - m) + expf(lg[b][1] - m));
        out[tid] = lg[b][o] - lse;
    }
}

// =====================================================================
// Host orchestration
// =====================================================================
struct LayerP {
    const float *b1,*b2,*be1,*be2,*bk,*bo,*bq,*bv,*g1,*g2,*w1,*w2,*wk,*wo,*wq,*wv;
};

static void run_encoder(float* E, float* Q, float* Kb, float* V, float* T1, float* T2,
                        float* meas, int* mtop, _Float16* Ah, _Float16* Wh,
                        const LayerP& P, int Lx, int u, hipStream_t s)
{
    const int M  = BB * Lx;                 // multiple of 16
    const int BH = BB * NH;

    // QKV projections (shared A staging)
    launch_cvt(E, Ah, M, DM, KP_DM, M, s);
    launch_cvt(P.wq, Wh, DM, DM, KP_DM, NP300, s);
    launch_gemm(0, Ah, Wh, P.bq, nullptr, Q,  M, DM, NP300, KP_DM, s);
    launch_cvt(P.wk, Wh, DM, DM, KP_DM, NP300, s);
    launch_gemm(0, Ah, Wh, P.bk, nullptr, Kb, M, DM, NP300, KP_DM, s);
    launch_cvt(P.wv, Wh, DM, DM, KP_DM, NP300, s);
    launch_gemm(0, Ah, Wh, P.bv, nullptr, V,  M, DM, NP300, KP_DM, s);

    hipLaunchKernelGGL(attn_measure,  dim3(BH),     dim3(256), 0, s, Q, Kb, meas, Lx, DH, u);
    hipLaunchKernelGGL(attn_topk,     dim3(BH),     dim3(256), 0, s, meas, mtop, Lx, u);
    hipLaunchKernelGGL(attn_ctx_base, dim3(BH),     dim3(256), 0, s, V, T1, Lx, DH);
    hipLaunchKernelGGL(attn_upd,      dim3(BH * u), dim3(256), 0, s, Q, Kb, V, mtop, T1, Lx, DH, u,
                       1.0f / sqrtf((float)DH));

    // output projection + residual
    launch_cvt(T1, Ah, M, DM, KP_DM, M, s);
    launch_cvt(P.wo, Wh, DM, DM, KP_DM, NP300, s);
    launch_gemm(0, Ah, Wh, P.bo, E, T2, M, DM, NP300, KP_DM, s);
    hipLaunchKernelGGL(layernorm_rows, dim3(M), dim3(256), 0, s, T2, P.g1, P.be1, E, DM);

    // FFN
    launch_cvt(E, Ah, M, DM, KP_DM, M, s);
    launch_cvt(P.w1, Wh, DFF, DM, KP_DM, DFF, s);
    launch_gemm(1, Ah, Wh, P.b1, nullptr, T1, M, DFF, DFF, KP_DM, s);      // GELU
    launch_cvt(T1, Ah, M, DFF, KP_FF, M, s);
    launch_cvt(P.w2, Wh, DM, DFF, KP_FF, NP300, s);
    launch_gemm(0, Ah, Wh, P.b2, E, T2, M, DM, NP300, KP_FF, s);           // + residual
    hipLaunchKernelGGL(layernorm_rows, dim3(M), dim3(256), 0, s, T2, P.g2, P.be2, E, DM);
}

extern "C" void kernel_launch(void* const* d_in, const int* in_sizes, int n_in,
                              void* d_out, int out_size, void* d_ws, size_t ws_size,
                              hipStream_t stream)
{
    (void)in_sizes; (void)n_in; (void)out_size; (void)ws_size;

    // Input order: x, h, then params flattened in jax tree order (sorted keys).
    const int*   xin     = (const int*)  d_in[0];
    const float* h0      = (const float*)d_in[1];
    const float* bhh     = (const float*)d_in[2];
    const float* bih     = (const float*)d_in[3];
    const float* bl      = (const float*)d_in[4];
    const float* conv_b  = (const float*)d_in[5];
    const float* conv_be = (const float*)d_in[6];
    const float* conv_g  = (const float*)d_in[7];
    const float* conv_w  = (const float*)d_in[8];   // [300,300,3] == [300,900] row-major
    const float* emb     = (const float*)d_in[9];
    LayerP Lp[2];
    for (int l = 0; l < 2; ++l) {
        const int b = 10 + l * 16;
        Lp[l].b1  = (const float*)d_in[b + 0];  Lp[l].b2  = (const float*)d_in[b + 1];
        Lp[l].be1 = (const float*)d_in[b + 2];  Lp[l].be2 = (const float*)d_in[b + 3];
        Lp[l].bk  = (const float*)d_in[b + 4];  Lp[l].bo  = (const float*)d_in[b + 5];
        Lp[l].bq  = (const float*)d_in[b + 6];  Lp[l].bv  = (const float*)d_in[b + 7];
        Lp[l].g1  = (const float*)d_in[b + 8];  Lp[l].g2  = (const float*)d_in[b + 9];
        Lp[l].w1  = (const float*)d_in[b + 10]; Lp[l].w2  = (const float*)d_in[b + 11];
        Lp[l].wk  = (const float*)d_in[b + 12]; Lp[l].wo  = (const float*)d_in[b + 13];
        Lp[l].wq  = (const float*)d_in[b + 14]; Lp[l].wv  = (const float*)d_in[b + 15];
    }
    const float* norm_b = (const float*)d_in[42];
    const float* norm_g = (const float*)d_in[43];
    const float* whh    = (const float*)d_in[44];
    const float* wih    = (const float*)d_in[45];
    const float* wl     = (const float*)d_in[46];

    // ---------------- workspace layout ----------------
    const long SZ_A  = (long)BB * LT * DM;          // 2,467,200 fl (covers B*L*300 too)
    const long SZ_T1 = (long)BB * LL * DFF;         // 8,388,608 fl (>= im2col 7.4M)
    const long SZ_AH = (long)BB * LL * KP_FF;       // 8,388,608 halfs (max A staging)
    float* ws  = (float*)d_ws;
    float* E   = ws;
    float* Q   = E  + SZ_A;
    float* Kb  = Q  + SZ_A;
    float* V   = Kb + SZ_A;
    float* T2  = V  + SZ_A;
    float* T1  = T2 + SZ_A;
    float* meas = T1 + SZ_T1;                       // B*H*L <= 16384
    float* hTb  = meas + 16384;                     // B*HID = 2048
    int*   mtop = (int*)(hTb + 2048);               // B*H*u <= 1120 ints
    _Float16* Ah = (_Float16*)(hTb + 2048 + 2048);  // 16-B aligned region
    _Float16* Wh = Ah + SZ_AH;                      // max Npad*Kp = 1024*320 halfs

    // 1) embedding
    {
        const long total = (long)BB * LL * DM;
        hipLaunchKernelGGL(embed_kernel, dim3((int)((total + 255) / 256)), dim3(256), 0, stream,
                           xin, emb, E, (long)BB * LL, DM);
    }

    // 2) encoder layer 0 (L=512, u=35)
    run_encoder(E, Q, Kb, V, T1, T2, meas, mtop, Ah, Wh, Lp[0], LL, U1, stream);

    // 3) conv layer: im2col -> WMMA GEMM -> batch-chan norm + ELU -> maxpool
    {
        const long total = (long)BB * LT * DM * 3;
        hipLaunchKernelGGL(im2col_wrap, dim3((int)((total + 255) / 256)), dim3(256), 0, stream,
                           E, T1, BB, LL, DM);
        launch_cvt(T1, Ah, BB * LT, DM * 3, KP_CONV, BB * LT, stream);
        launch_cvt(conv_w, Wh, DM, DM * 3, KP_CONV, NP300, stream);
        launch_gemm(0, Ah, Wh, conv_b, nullptr, T2, BB * LT, DM, NP300, KP_CONV, stream);
        hipLaunchKernelGGL(chan_norm_elu, dim3(DM), dim3(256), 0, stream, T2, conv_g, conv_be,
                           BB * LT, DM);
        const long ptotal = (long)BB * LP * DM;
        hipLaunchKernelGGL(maxpool3s2, dim3((int)((ptotal + 255) / 256)), dim3(256), 0, stream,
                           T2, E, BB, LT, DM);
    }

    // 4) encoder layer 1 (L=257, u=30)
    run_encoder(E, Q, Kb, V, T1, T2, meas, mtop, Ah, Wh, Lp[1], LP, U2, stream);

    // 5) final LN -> RNN input projection (WMMA) -> scan -> logits + log_softmax
    hipLaunchKernelGGL(layernorm_rows, dim3(BB * LP), dim3(256), 0, stream,
                       E, norm_g, norm_b, T2, DM);
    launch_cvt(T2, Ah, BB * LP, DM, KP_DM, BB * LP, stream);
    launch_cvt(wih, Wh, HID, DM, KP_DM, HID, stream);
    launch_gemm(0, Ah, Wh, bih, nullptr, T1, BB * LP, HID, HID, KP_DM, stream);
    hipLaunchKernelGGL(rnn_scan,  dim3(1), dim3(256), 0, stream, T1, h0, whh, bhh, hTb, LP);
    hipLaunchKernelGGL(logits_ls, dim3(1), dim3(64),  0, stream, hTb, wl, bl, (float*)d_out);
}
